// ShiftedWindowAttention_4715874091876
// MI455X (gfx1250) — compile-verified, matrix operand order checked by reading, not run on hardware
//
#include <hip/hip_runtime.h>
#include <hip/hip_bf16.h>
#include <cstddef>

typedef __attribute__((ext_vector_type(16))) _Float16 v16h;
typedef __attribute__((ext_vector_type(4)))  _Float16 v4h;
typedef __attribute__((ext_vector_type(8)))  float    v8f;

#define NWIN  64
#define NTOK  512
#define CDIM  128
#define NHEAD 4
#define HD    32
#define VT_LD (NTOK + 8)   // padded leading dim for transposed V (bank spread)

// ---------------------------------------------------------------------------
// WMMA fragment helper (wave32, 16x16x32 f16 -> f32)
// A/B fragment: element (i, j): lane<16 -> i=lane, j in {0..7,16..23};
//               lane>=16 -> i=lane-16, j in {8..15,24..31}
// Loaded from row-major [16][K] tile with leading dim ld.
// Contiguous 8-half groups -> lowers to ds_load_b128 / global_load_b128.
// ---------------------------------------------------------------------------
__device__ __forceinline__ v16h load_frag(const _Float16* src, int ld) {
    int lane = threadIdx.x & 31;
    int row  = lane & 15;
    int k0   = (lane < 16) ? 0 : 8;
    const _Float16* p = src + row * ld;
    v16h f;
#pragma unroll
    for (int t = 0; t < 8; ++t) {
        f[t]     = p[k0 + t];
        f[t + 8] = p[k0 + 16 + t];
    }
    return f;
}

__device__ __forceinline__ v8f wmma_f16(v16h a, v16h b, v8f c) {
    return __builtin_amdgcn_wmma_f32_16x16x32_f16(false, a, false, b,
                                                  (short)0, c, false, false);
}

// region label along one axis for the shift mask (H=32, win=8, shift=4)
__device__ __forceinline__ int axis_label(int c) {
    return (c < 24) ? 0 : ((c < 28) ? 1 : 2);
}

// ---------------------------------------------------------------------------
// Kernel 0: materialize relative-position bias  bias[h][i][j], h in [0,4)
// (4 MB total -> stays resident in the 192 MB L2 for kernel 2)
// ---------------------------------------------------------------------------
__global__ void swa_bias_gather(const float* __restrict__ tbl,
                                const int*   __restrict__ rpi,
                                float*       __restrict__ bias_ws) {
    int idx = blockIdx.x * blockDim.x + threadIdx.x;     // over N*N
    if (idx < NTOK * NTOK) {
        int t = rpi[idx];
#pragma unroll
        for (int h = 0; h < NHEAD; ++h)
            bias_ws[(size_t)h * NTOK * NTOK + idx] = tbl[t * NHEAD + h];
    }
}

// ---------------------------------------------------------------------------
// Kernel 1: fused roll(-4) + window partition + QKV GEMM (WMMA), f16 out
// grid = 64 windows, block = 256 (8 waves)
// qws/kws/vws layout: [win][head][tok][32] f16
// ---------------------------------------------------------------------------
__global__ void swa_qkv_win(const float* __restrict__ x,
                            const float* __restrict__ w,   // [384][128]
                            const float* __restrict__ b,   // [384]
                            _Float16* __restrict__ qws,
                            _Float16* __restrict__ kws,
                            _Float16* __restrict__ vws) {
    extern __shared__ char smem[];
    _Float16* xw = (_Float16*)smem;                        // [512][128]
    _Float16* wq = xw + NTOK * CDIM;                       // [384][128]

    int win = blockIdx.x;
    int wh = win >> 4, ww = (win >> 2) & 3, wd = win & 3;
    int tid = threadIdx.x, wave = tid >> 5, lane = tid & 31;

    for (int i = tid; i < 3 * CDIM * CDIM; i += 256)
        wq[i] = (_Float16)w[i];

    for (int tok = wave; tok < NTOK; tok += 8) {
        int ih = tok >> 6, iw = (tok >> 3) & 7, id = tok & 7;
        int h = (wh * 8 + ih + 4) & 31;
        int wc = (ww * 8 + iw + 4) & 31;
        int d = (wd * 8 + id + 4) & 31;
        const float* src = x + (((size_t)h * 32 + wc) * 32 + d) * CDIM;
        // hide gather latency of the cyclic shift (global_prefetch_b8)
        __builtin_prefetch(src + 8 * CDIM, 0, 0);
        for (int c = lane; c < CDIM; c += 32)
            xw[tok * CDIM + c] = (_Float16)src[c];
    }
    __syncthreads();

    const float qscale = 0.17677669529663689f;             // 32^-0.5
    // 32 row tiles x 24 col tiles
    for (int tile = wave; tile < 32 * 24; tile += 8) {
        int rt = tile / 24, ct = tile % 24;
        v8f acc = {};
#pragma unroll
        for (int kk = 0; kk < 4; ++kk) {
            v16h A = load_frag(xw + rt * 16 * CDIM + kk * 32, CDIM);
            v16h B = load_frag(wq + ct * 16 * CDIM + kk * 32, CDIM);
            acc = wmma_f16(A, B, acc);
        }
        int n  = lane & 15;
        int m0 = (lane < 16) ? 0 : 8;
        int c  = ct * 16 + n;
        float bias = b[c];
        int kind = c >> 7;                                  // 0=q 1=k 2=v
        int cc = c & (CDIM - 1);
        int head = cc >> 5, hd = cc & 31;
        _Float16* dst = (kind == 0) ? qws : ((kind == 1) ? kws : vws);
        float scl = (kind == 0) ? qscale : 1.0f;
        size_t base = (((size_t)win * NHEAD + head) * NTOK) * HD + hd;
#pragma unroll
        for (int r = 0; r < 8; ++r) {
            int tok = rt * 16 + m0 + r;
            dst[base + (size_t)tok * HD] = (_Float16)((acc[r] + bias) * scl);
        }
    }
}

// ---------------------------------------------------------------------------
// Kernel 2: attention per (win, head). grid = 256, block = 128 (4 waves)
// K row-major and V TRANSPOSED (padded) in LDS; each wave streams 16-row
// strips: WMMA QK^T (+bias+mask), vectorized 2-lane/row softmax, f16 P,
// WMMA P*V (B fragment now contiguous -> ds_load_b128).
// Output ows: [win][tok][head*32+hd] f32
// ---------------------------------------------------------------------------
__global__ void swa_attn(const _Float16* __restrict__ qws,
                         const _Float16* __restrict__ kws,
                         const _Float16* __restrict__ vws,
                         const float*    __restrict__ bias_ws,
                         float*          __restrict__ ows) {
    extern __shared__ char smem[];
    _Float16* Ks = (_Float16*)smem;                        // [512][32]
    _Float16* Vt = Ks + NTOK * HD;                         // [32][VT_LD] transposed
    float*    Ss = (float*)(Vt + HD * VT_LD);              // [4][16][512]
    _Float16* Ps = (_Float16*)(Ss + 4 * 16 * NTOK);        // [4][16][512]
    unsigned char* lab = (unsigned char*)(Ps + 4 * 16 * NTOK); // [512]

    int blk = blockIdx.x;
    int win = blk >> 2, head = blk & 3;
    int wh = win >> 4, ww = (win >> 2) & 3, wd = win & 3;
    int tid = threadIdx.x, wave = tid >> 5, lane = tid & 31;

    size_t base = (((size_t)win * NHEAD + head) * NTOK) * HD;
    const _Float16* qg = qws + base;
    for (int i = tid; i < NTOK * HD; i += 128) {
        Ks[i] = kws[base + i];                             // K row-major
        int tok = i >> 5, hd = i & 31;
        Vt[hd * VT_LD + tok] = vws[base + i];              // V transposed
    }
    for (int t = tid; t < NTOK; t += 128) {
        int ih = t >> 6, iw = (t >> 3) & 7, id = t & 7;
        int la = axis_label(wh * 8 + ih);
        int lb = axis_label(ww * 8 + iw);
        int lc = axis_label(wd * 8 + id);
        lab[t] = (unsigned char)(la * 9 + lb * 3 + lc);
    }
    __syncthreads();

    float*    Sst = Ss + wave * 16 * NTOK;
    _Float16* Pst = Ps + wave * 16 * NTOK;
    const float* bh = bias_ws + (size_t)head * NTOK * NTOK;

    // 32 strips, 8 per wave -> uniform barrier count across waves
    for (int strip = wave; strip < 32; strip += 4) {
        int i0 = strip * 16;
        v16h A = load_frag(qg + i0 * HD, HD);              // Q strip, K=32=full hd
        for (int ct = 0; ct < 32; ++ct) {
            v16h B = load_frag(Ks + ct * 16 * HD, HD);     // K^T tile
            v8f acc = {};
            acc = wmma_f16(A, B, acc);
            int n  = lane & 15;
            int m0 = (lane < 16) ? 0 : 8;
            int j  = ct * 16 + n;
            unsigned char lj = lab[j];
#pragma unroll
            for (int r = 0; r < 8; ++r) {
                int i = m0 + r;
                float v = acc[r] + bh[(size_t)(i0 + i) * NTOK + j];
                if (lab[i0 + i] != lj) v -= 100.0f;
                Sst[i * NTOK + j] = v;
            }
        }
        __syncthreads();

        // softmax: 2 lanes per row, float4-vectorized, pair-combine via shfl
        {
            int r  = lane & 15;                            // row in strip
            int hs = lane >> 4;                            // which half of row
            float* row = Sst + r * NTOK;
            float4* rv = (float4*)row + hs * 64;           // 64 float4 per lane
            float mx = -1e30f;
            for (int t = 0; t < 64; ++t) {
                float4 v = rv[t];
                mx = fmaxf(mx, fmaxf(fmaxf(v.x, v.y), fmaxf(v.z, v.w)));
            }
            mx = fmaxf(mx, __shfl_xor(mx, 16, 32));
            float sum = 0.0f;
            for (int t = 0; t < 64; ++t) {
                float4 v = rv[t];
                v.x = __expf(v.x - mx); v.y = __expf(v.y - mx);
                v.z = __expf(v.z - mx); v.w = __expf(v.w - mx);
                rv[t] = v;
                sum += (v.x + v.y) + (v.z + v.w);
            }
            sum += __shfl_xor(sum, 16, 32);
            float inv = 1.0f / sum;
            v4h* prow = (v4h*)(Pst + r * NTOK) + hs * 64;
            for (int t = 0; t < 64; ++t) {
                float4 v = rv[t];
                v4h p;
                p[0] = (_Float16)(v.x * inv); p[1] = (_Float16)(v.y * inv);
                p[2] = (_Float16)(v.z * inv); p[3] = (_Float16)(v.w * inv);
                prow[t] = p;
            }
        }
        __syncthreads();

        // O strip = P(16x512) @ V(512x32): 2 col tiles, 16 K-steps each
#pragma unroll
        for (int ct = 0; ct < 2; ++ct) {
            v8f acc = {};
            for (int kk = 0; kk < 16; ++kk) {
                v16h Ap = load_frag(Pst + kk * 32, NTOK);
                // B element (n, k) = V[k][ct*16+n] = Vt[(ct*16+n)*VT_LD + k]
                v16h Bv = load_frag(Vt + (ct * 16) * VT_LD + kk * 32, VT_LD);
                acc = wmma_f16(Ap, Bv, acc);
            }
            int n  = lane & 15;
            int m0 = (lane < 16) ? 0 : 8;
#pragma unroll
            for (int r = 0; r < 8; ++r) {
                int i = i0 + m0 + r;
                ows[((size_t)win * NTOK + i) * CDIM + head * HD + ct * 16 + n] = acc[r];
            }
        }
        __syncthreads();
    }
}

// ---------------------------------------------------------------------------
// Kernel 3: proj GEMM + bias + reverse window partition + roll(+4)
// grid = 64 windows, block = 256 (8 waves)
// ---------------------------------------------------------------------------
__global__ void swa_proj_win(const float* __restrict__ ows,
                             const float* __restrict__ pw,  // [128][128]
                             const float* __restrict__ pb,  // [128]
                             float* __restrict__ out) {
    extern __shared__ char smem[];
    _Float16* ow = (_Float16*)smem;                        // [512][128]
    _Float16* wp = ow + NTOK * CDIM;                       // [128][128]

    int win = blockIdx.x;
    int wh = win >> 4, ww = (win >> 2) & 3, wd = win & 3;
    int tid = threadIdx.x, wave = tid >> 5, lane = tid & 31;

    for (int i = tid; i < CDIM * CDIM; i += 256)
        wp[i] = (_Float16)pw[i];
    for (int i = tid; i < NTOK * CDIM; i += 256)
        ow[i] = (_Float16)ows[(size_t)win * NTOK * CDIM + i];
    __syncthreads();

    // 32 row tiles x 8 col tiles
    for (int tile = wave; tile < 32 * 8; tile += 8) {
        int rt = tile >> 3, ct = tile & 7;
        v8f acc = {};
#pragma unroll
        for (int kk = 0; kk < 4; ++kk) {
            v16h A = load_frag(ow + rt * 16 * CDIM + kk * 32, CDIM);
            v16h B = load_frag(wp + ct * 16 * CDIM + kk * 32, CDIM);
            acc = wmma_f16(A, B, acc);
        }
        int n  = lane & 15;
        int m0 = (lane < 16) ? 0 : 8;
        int c  = ct * 16 + n;
        float bias = pb[c];
#pragma unroll
        for (int r = 0; r < 8; ++r) {
            int tok = rt * 16 + m0 + r;
            int ih = tok >> 6, iw = (tok >> 3) & 7, id = tok & 7;
            int h  = (wh * 8 + ih + 4) & 31;
            int w2 = (ww * 8 + iw + 4) & 31;
            int d2 = (wd * 8 + id + 4) & 31;
            out[(((size_t)h * 32 + w2) * 32 + d2) * CDIM + c] = acc[r] + bias;
        }
    }
}

// ---------------------------------------------------------------------------
// Host launcher
// ---------------------------------------------------------------------------
extern "C" void kernel_launch(void* const* d_in, const int* in_sizes, int n_in,
                              void* d_out, int out_size, void* d_ws, size_t ws_size,
                              hipStream_t stream) {
    const float* x   = (const float*)d_in[0];
    const float* qw  = (const float*)d_in[1];
    const float* qb  = (const float*)d_in[2];
    const float* pw  = (const float*)d_in[3];
    const float* pb  = (const float*)d_in[4];
    const float* tbl = (const float*)d_in[5];
    const int*   rpi = (const int*)d_in[6];
    float* out = (float*)d_out;

    // workspace partition
    char* ws = (char*)d_ws;
    size_t qkv_bytes = (size_t)NWIN * NHEAD * NTOK * HD * 2;   // 8 MB each (f16)
    _Float16* qws = (_Float16*)ws;                  ws += qkv_bytes;
    _Float16* kws = (_Float16*)ws;                  ws += qkv_bytes;
    _Float16* vws = (_Float16*)ws;                  ws += qkv_bytes;
    float* bias_ws = (float*)ws;                    ws += (size_t)NHEAD * NTOK * NTOK * 4; // 4 MB
    float* ows = (float*)ws;                        // 16 MB

    const int SM_QKV  = NTOK * CDIM * 2 + 3 * CDIM * CDIM * 2;            // 224 KB
    const int SM_ATTN = NTOK * HD * 2 + HD * VT_LD * 2
                      + 4 * 16 * NTOK * 4 + 4 * 16 * NTOK * 2 + NTOK;     // 257 KB
    const int SM_PROJ = NTOK * CDIM * 2 + CDIM * CDIM * 2;                 // 160 KB

    hipFuncSetAttribute((const void*)swa_qkv_win,
                        hipFuncAttributeMaxDynamicSharedMemorySize, SM_QKV);
    hipFuncSetAttribute((const void*)swa_attn,
                        hipFuncAttributeMaxDynamicSharedMemorySize, SM_ATTN);
    hipFuncSetAttribute((const void*)swa_proj_win,
                        hipFuncAttributeMaxDynamicSharedMemorySize, SM_PROJ);

    swa_bias_gather<<<(NTOK * NTOK + 255) / 256, 256, 0, stream>>>(tbl, rpi, bias_ws);
    swa_qkv_win<<<NWIN, 256, SM_QKV, stream>>>(x, qw, qb, qws, kws, vws);
    swa_attn<<<NWIN * NHEAD, 128, SM_ATTN, stream>>>(qws, kws, vws, bias_ws, ows);
    swa_proj_win<<<NWIN, 256, SM_PROJ, stream>>>(ows, pw, pb, out);
}
